// DynamicGRU_61022895341974
// MI455X (gfx1250) — matrix operand order repaired and checked by reading, not compiled, not run on hardware
//
#include <hip/hip_runtime.h>

#define Bb 4
#define Nn 50
#define Tt 48
#define Pp 10
#define Dd 256
#define TWOD 512
#define NROWS (Bb * Nn * Pp)            // 2000
#define ROWS_PER_BLK 32                 // 2 M-tiles per block
#define NBLK ((NROWS + ROWS_PER_BLK - 1) / ROWS_PER_BLK)   // 63 (last partial)
#define XH_STRIDE 520                   // 512 + 8 bf16 pad (16B/row bank shift)
#define F_STRIDE 260                    // 256 + 4 f32 pad
#define R_STRIDE 264                    // 256 + 8 bf16 pad
#define LN_EPSF 1e-5f
#define ZR 0.1f

// ---- manual LDS pool layout: xh_s at offset 0 so hot-loop DS addresses
// ---- fit the 16-bit immediate-offset field -------------------------------
#define XH_OFF   0
#define XH_BYTES (ROWS_PER_BLK * XH_STRIDE * 2)   // 33280
#define H_OFF    (XH_OFF + XH_BYTES)              // 33280
#define H_BYTES  (ROWS_PER_BLK * F_STRIDE * 4)    // 33280
#define R_OFF    (H_OFF + H_BYTES)                // 66560
#define R_BYTES  (ROWS_PER_BLK * R_STRIDE * 2)    // 16896
#define BIAS_OFF (R_OFF + R_BYTES)                // 83456
#define GB_OFF   (BIAS_OFF + 3 * Dd * 4)          // 86528
#define U_OFF    (GB_OFF + 2 * Dd * 4)            // 88576
#define HH_OFF   (U_OFF + H_BYTES)                // 121856
#define LDS_TOTAL (HH_OFF + H_BYTES)              // 155136

typedef __attribute__((ext_vector_type(8)))  __bf16 v8bf;
typedef __attribute__((ext_vector_type(16))) __bf16 v16bf;
typedef __attribute__((ext_vector_type(8)))  float  v8f;

__device__ __forceinline__ float fast_sigmoid(float v) {
  return __builtin_amdgcn_rcpf(1.0f + __expf(-v));
}
__device__ __forceinline__ float fast_tanh(float v) {
  return 1.0f - 2.0f * __builtin_amdgcn_rcpf(1.0f + __expf(2.0f * v));
}

// ---------------------------------------------------------------------------
// Pre-swizzle fp32 weights [512][256] into bf16 WMMA B-fragments.
//   idx = (((g*16 + nt)*16 + kt)*32 + l)*16 + e
//   column n = nt*16 + (l&15);  k = kt*32 + (l>>4)*8 + e (e<8), kt*32 + 16 + (l>>4)*8 + (e-8) (e>=8)
// ---------------------------------------------------------------------------
__global__ __launch_bounds__(256) void prep_weights(
    const float* __restrict__ Wr, const float* __restrict__ Wu,
    const float* __restrict__ Wh, __bf16* __restrict__ wf) {
  int idx = blockIdx.x * 256 + threadIdx.x;     // 0 .. 3*512*256-1
  int e  = idx & 15;
  int l  = (idx >> 4) & 31;
  int kt = (idx >> 9) & 15;
  int nt = (idx >> 13) & 15;
  int g  = idx >> 17;
  int n  = nt * 16 + (l & 15);
  int k  = kt * 32 + (l >> 4) * 8 + (e & 7) + (e >> 3) * 16;
  const float* W = (g == 0) ? Wr : (g == 1) ? Wu : Wh;
  wf[idx] = (__bf16)W[k * Dd + n];
}

// A-fragment load; rowp pre-offset by half*8 so loads are rowp + kt*32 (+16).
__device__ __forceinline__ v16bf load_afrag(const __bf16* rowp, int kt) {
  v8bf lo = *(const v8bf*)(rowp + kt * 32);
  v8bf hi = *(const v8bf*)(rowp + kt * 32 + 16);
  v16bf a;
#pragma unroll
  for (int e = 0; e < 8; ++e) { a[e] = lo[e]; a[8 + e] = hi[e]; }
  return a;
}

// ---------------------------------------------------------------------------
// Persistent per-block GRU scan: 32 rows (2 M-tiles) per block.
// ---------------------------------------------------------------------------
__global__ __launch_bounds__(256) void gru_kernel(
    const float* __restrict__ x, const __bf16* __restrict__ wf,
    const float* __restrict__ br, const float* __restrict__ bu,
    const float* __restrict__ bh, const float* __restrict__ gamma,
    const float* __restrict__ beta, float* __restrict__ out) {
  __shared__ __align__(16) unsigned char lds_pool[LDS_TOTAL];
  __bf16 (*xh_s)[XH_STRIDE] = (__bf16(*)[XH_STRIDE])(lds_pool + XH_OFF);
  float  (*h_s)[F_STRIDE]   = (float(*)[F_STRIDE])(lds_pool + H_OFF);
  __bf16 (*r_s)[R_STRIDE]   = (__bf16(*)[R_STRIDE])(lds_pool + R_OFF);
  float  (*bias_s)[Dd]      = (float(*)[Dd])(lds_pool + BIAS_OFF);
  float  (*gb_s)[Dd]        = (float(*)[Dd])(lds_pool + GB_OFF);
  float  (*u_s)[F_STRIDE]   = (float(*)[F_STRIDE])(lds_pool + U_OFF);
  float  (*hh_s)[F_STRIDE]  = (float(*)[F_STRIDE])(lds_pool + HH_OFF);

  const int tid  = threadIdx.x;
  const int w    = tid >> 5;      // wave id 0..7
  const int l    = tid & 31;      // lane
  const int ln   = l & 15;        // row m (A) / col n (B,C)
  const int half = l >> 4;

  const int rrow = tid >> 4;      // 0..15 ; handles rows rrow and rrow+16
  const int rl   = tid & 15;

  // Preload biases, gamma, beta.
  bias_s[0][tid] = br[tid];
  bias_s[1][tid] = bu[tid];
  bias_s[2][tid] = bh[tid];
  gb_s[0][tid] = gamma[tid];
  gb_s[1][tid] = beta[tid];

  // Per-row-group global bases + validity (last block is partial).
  const long tstride = (long)Pp * Dd;
  const float* xbase[2];
  float*       obase[2];
  bool         valid[2];
#pragma unroll
  for (int rg = 0; rg < 2; ++rg) {
    const int grow = blockIdx.x * ROWS_PER_BLK + rrow + 16 * rg;
    valid[rg] = (grow < NROWS);
    const int gr  = valid[rg] ? grow : 0;
    const int b   = gr / (Nn * Pp);
    const int rem = gr % (Nn * Pp);
    const long rowbase =
        ((long)(b * Nn + rem / Pp) * Tt * Pp + rem % Pp) * Dd;
    xbase[rg] = x + rowbase;
    obase[rg] = out + rowbase;
  }

  // Init h = 0 and the bf16 h-half of xh (step 1 no longer writes it).
  const float4 z4 = make_float4(0.f, 0.f, 0.f, 0.f);
#pragma unroll
  for (int rg = 0; rg < 2; ++rg)
#pragma unroll
    for (int i = 0; i < 4; ++i) {
      const int c = rl * 4 + i * 64;
      const int rr = rrow + 16 * rg;
      *(float4*)&h_s[rr][c] = z4;
      __bf16* dh = &xh_s[rr][Dd + c];
      dh[0] = (__bf16)0.f; dh[1] = (__bf16)0.f;
      dh[2] = (__bf16)0.f; dh[3] = (__bf16)0.f;
    }

  __syncthreads();

  const __bf16* arow0 = &xh_s[ln][half * 8];        // A base, M-tile 0
  const __bf16* arow1 = &xh_s[16 + ln][half * 8];   // A base, M-tile 1

#pragma unroll 1
  for (int t = 0; t < Tt; ++t) {
    // ---- 1) build x-half of xh (h-half maintained by step 5) ---------------
#pragma unroll
    for (int rg = 0; rg < 2; ++rg) {
      const int rr = rrow + 16 * rg;
      const float* xr = xbase[rg] + t * tstride;
      if (t + 1 < Tt && rg == 0) __builtin_prefetch(xr + tstride, 0, 1);
#pragma unroll
      for (int i = 0; i < 4; ++i) {
        const int c = rl * 4 + i * 64;
        float4 xv = valid[rg] ? *(const float4*)(xr + c) : z4;
        __bf16* dx = &xh_s[rr][c];
        dx[0] = (__bf16)xv.x; dx[1] = (__bf16)xv.y;
        dx[2] = (__bf16)xv.z; dx[3] = (__bf16)xv.w;
      }
    }
    __syncthreads();

    // ---- 2) phase 1: r,u gates. 32 jobs (2 gates x 16 ntiles) / 8 waves.
    //      wbase/bias/col computed inline: keeps wf provenance = global
    //      address space (global_load_b128, no flat path, no cndmask chains).
#pragma unroll 1
    for (int jj = 0; jj < 4; ++jj) {
      const int job = w + jj * 8;
      const int g   = job >> 4;     // 0 = r, 1 = u
      const int nt  = job & 15;
      v8f acc0 = {}, acc1 = {};
      const __bf16* wbase = wf + ((g * 16 + nt) * 16) * 512 + l * 16;
#pragma unroll 4
      for (int kt = 0; kt < 16; ++kt) {
        v16bf bfrag = *(const v16bf*)(wbase + kt * 512);
        v16bf a0 = load_afrag(arow0, kt);
        v16bf a1 = load_afrag(arow1, kt);
        acc0 = __builtin_amdgcn_wmma_f32_16x16x32_bf16(
            false, a0, false, bfrag, (short)0, acc0, false, false);
        acc1 = __builtin_amdgcn_wmma_f32_16x16x32_bf16(
            false, a1, false, bfrag, (short)0, acc1, false, false);
      }
      const int   col  = nt * 16 + ln;
      const float bias = bias_s[g][col];
#pragma unroll
      for (int v = 0; v < 8; ++v) {
        const int row = v + 8 * half;
        float s0 = fast_sigmoid(acc0[v] + bias);
        float s1 = fast_sigmoid(acc1[v] + bias);
        if (g) { u_s[row][col] = s0;         u_s[16 + row][col] = s1; }
        else   { r_s[row][col] = (__bf16)s0; r_s[16 + row][col] = (__bf16)s1; }
      }
    }
    __syncthreads();

    // ---- 3) rebuild h-half of xh as bf16(h * r) ----------------------------
#pragma unroll
    for (int rg = 0; rg < 2; ++rg) {
      const int rr = rrow + 16 * rg;
#pragma unroll
      for (int i = 0; i < 4; ++i) {
        const int c = rl * 4 + i * 64;
        float4 hv = *(const float4*)&h_s[rr][c];
        __bf16* dh = &xh_s[rr][Dd + c];
        dh[0] = (__bf16)(hv.x * (float)r_s[rr][c + 0]);
        dh[1] = (__bf16)(hv.y * (float)r_s[rr][c + 1]);
        dh[2] = (__bf16)(hv.z * (float)r_s[rr][c + 2]);
        dh[3] = (__bf16)(hv.w * (float)r_s[rr][c + 3]);
      }
    }
    __syncthreads();

    // ---- 4) phase 2: h_hat. 16 ntile jobs / 8 waves, both M-tiles each. ----
#pragma unroll 1
    for (int jj = 0; jj < 2; ++jj) {
      const int nt = w + jj * 8;
      v8f acc0 = {}, acc1 = {};
      const __bf16* wbase = wf + ((2 * 16 + nt) * 16) * 512 + l * 16;
#pragma unroll 4
      for (int kt = 0; kt < 16; ++kt) {
        v16bf bfrag = *(const v16bf*)(wbase + kt * 512);
        v16bf a0 = load_afrag(arow0, kt);
        v16bf a1 = load_afrag(arow1, kt);
        acc0 = __builtin_amdgcn_wmma_f32_16x16x32_bf16(
            false, a0, false, bfrag, (short)0, acc0, false, false);
        acc1 = __builtin_amdgcn_wmma_f32_16x16x32_bf16(
            false, a1, false, bfrag, (short)0, acc1, false, false);
      }
      const int   col  = nt * 16 + ln;
      const float bias = bias_s[2][col];
#pragma unroll
      for (int v = 0; v < 8; ++v) {
        const int row = v + 8 * half;
        hh_s[row][col]      = fast_tanh(acc0[v] + bias);
        hh_s[16 + row][col] = fast_tanh(acc1[v] + bias);
      }
    }
    __syncthreads();

    // ---- 5) fused gate-mix + layernorm + zoneout + state/out stores --------
#pragma unroll
    for (int rg = 0; rg < 2; ++rg) {
      const int rr = rrow + 16 * rg;
      float4 pre[4], hold[4];
      float sum = 0.f, sumsq = 0.f;
#pragma unroll
      for (int i = 0; i < 4; ++i) {
        const int c = rl * 4 + i * 64;
        float4 hv = *(const float4*)&h_s[rr][c];
        float4 uv = *(const float4*)&u_s[rr][c];
        float4 hh = *(const float4*)&hh_s[rr][c];
        float4 pv;
        pv.x = fmaf(uv.x, hh.x - hv.x, hv.x);
        pv.y = fmaf(uv.y, hh.y - hv.y, hv.y);
        pv.z = fmaf(uv.z, hh.z - hv.z, hv.z);
        pv.w = fmaf(uv.w, hh.w - hv.w, hv.w);
        pre[i] = pv; hold[i] = hv;
        sum   += pv.x + pv.y + pv.z + pv.w;
        sumsq += pv.x * pv.x + pv.y * pv.y + pv.z * pv.z + pv.w * pv.w;
      }
#pragma unroll
      for (int off = 8; off >= 1; off >>= 1) {
        sum   += __shfl_xor(sum, off, 16);
        sumsq += __shfl_xor(sumsq, off, 16);
      }
      const float mu   = sum * (1.0f / Dd);
      const float var  = sumsq * (1.0f / Dd) - mu * mu;
      const float rstd = rsqrtf(var + LN_EPSF);
      float* orow = obase[rg] + t * tstride;
#pragma unroll
      for (int i = 0; i < 4; ++i) {
        const int c = rl * 4 + i * 64;
        float4 g4 = *(const float4*)&gb_s[0][c];
        float4 b4 = *(const float4*)&gb_s[1][c];
        float4 pv = pre[i], hv = hold[i], o;
        o.x = ZR * hv.x + (1.0f - ZR) * fmaf((pv.x - mu) * rstd, g4.x, b4.x);
        o.y = ZR * hv.y + (1.0f - ZR) * fmaf((pv.y - mu) * rstd, g4.y, b4.y);
        o.z = ZR * hv.z + (1.0f - ZR) * fmaf((pv.z - mu) * rstd, g4.z, b4.z);
        o.w = ZR * hv.w + (1.0f - ZR) * fmaf((pv.w - mu) * rstd, g4.w, b4.w);
        *(float4*)&h_s[rr][c] = o;                 // next-step state (f32)
        __bf16* dh = &xh_s[rr][Dd + c];            // next-step A operand (bf16)
        dh[0] = (__bf16)o.x; dh[1] = (__bf16)o.y;
        dh[2] = (__bf16)o.z; dh[3] = (__bf16)o.w;
        if (valid[rg]) *(float4*)(orow + c) = o;   // output
      }
    }
    __syncthreads();
  }
}

extern "C" void kernel_launch(void* const* d_in, const int* in_sizes, int n_in,
                              void* d_out, int out_size, void* d_ws, size_t ws_size,
                              hipStream_t stream) {
  const float* x   = (const float*)d_in[0];
  const float* Wr  = (const float*)d_in[1];
  const float* brp = (const float*)d_in[2];
  const float* Wu  = (const float*)d_in[3];
  const float* bup = (const float*)d_in[4];
  const float* Wh  = (const float*)d_in[5];
  const float* bhp = (const float*)d_in[6];
  const float* gam = (const float*)d_in[7];
  const float* bet = (const float*)d_in[8];
  __bf16* wf = (__bf16*)d_ws;          // 3*512*256 bf16 = 768 KB

  prep_weights<<<(3 * TWOD * Dd) / 256, 256, 0, stream>>>(Wr, Wu, Wh, wf);
  gru_kernel<<<NBLK, 256, 0, stream>>>(x, wf, brp, bup, bhp, gam, bet,
                                       (float*)d_out);
}
